// SAMBA_Encoder_13314398618468
// MI455X (gfx1250) — compile-verified, matrix operand order checked by reading, not run on hardware
//
#include <hip/hip_runtime.h>
#include <cmath>

// Problem constants (match reference)
#define Bdim   8
#define Ldim   2048
#define Fdim   128
#define Hdim   256
#define DIN    512           // d_in = 2*H
#define DSTATE 16
#define DCONV  4
#define DTRANK 16
#define TOK    (Bdim * Ldim) // 16384 tokens
#define NC     16            // scan chunks
#define CL     (Ldim / NC)   // 128 steps per chunk

typedef __attribute__((ext_vector_type(16))) _Float16 v16h;
typedef __attribute__((ext_vector_type(8)))  float    v8f;

// ---------------------------------------------------------------------------
// WMMA 16x32 f16 fragment loaders (CDNA5 ISA 7.12.2 16-bit operand layout):
// lane row = lane%16, element e in [0,8) -> K = 8*(lane/16)+e,
// element e in [8,16) -> K = 16 + 8*(lane/16) + (e-8).
// Unguarded float4 loads (16B-aligned by construction) + f32->f16 convert.
// ---------------------------------------------------------------------------
__device__ __forceinline__ v16h load_frag32(const float* __restrict__ p, int hh) {
  const float4* q0 = reinterpret_cast<const float4*>(p + hh * 8);
  const float4* q1 = reinterpret_cast<const float4*>(p + 16 + hh * 8);
  float4 a0 = q0[0], a1 = q0[1];
  float4 b0 = q1[0], b1 = q1[1];
  v16h f;
  f[0]  = (_Float16)a0.x; f[1]  = (_Float16)a0.y; f[2]  = (_Float16)a0.z; f[3]  = (_Float16)a0.w;
  f[4]  = (_Float16)a1.x; f[5]  = (_Float16)a1.y; f[6]  = (_Float16)a1.z; f[7]  = (_Float16)a1.w;
  f[8]  = (_Float16)b0.x; f[9]  = (_Float16)b0.y; f[10] = (_Float16)b0.z; f[11] = (_Float16)b0.w;
  f[12] = (_Float16)b1.x; f[13] = (_Float16)b1.y; f[14] = (_Float16)b1.z; f[15] = (_Float16)b1.w;
  return f;
}

// K=16 variant: K-slots [16,32) are compile-time zero.
__device__ __forceinline__ v16h load_frag16(const float* __restrict__ p, int hh) {
  const float4* q0 = reinterpret_cast<const float4*>(p + hh * 8);
  float4 a0 = q0[0], a1 = q0[1];
  v16h f;
  f[0]  = (_Float16)a0.x; f[1]  = (_Float16)a0.y; f[2]  = (_Float16)a0.z; f[3]  = (_Float16)a0.w;
  f[4]  = (_Float16)a1.x; f[5]  = (_Float16)a1.y; f[6]  = (_Float16)a1.z; f[7]  = (_Float16)a1.w;
#pragma unroll
  for (int i = 8; i < 16; ++i) f[i] = (_Float16)0.0f;
  return f;
}

// ---------------------------------------------------------------------------
// Generic GEMM: out[m,n] = sum_k A[m,k] * W[n,k] (+bias[n]) (+postop)
// postop: 0 = none, 1 = softplus(v).  resid != nullptr -> v += resid[m*ldo+n].
// One wave computes a 16x(16*NT) strip: A fragment reused NT times per K-step,
// NT back-to-back v_wmma_f32_16x16x32_f16 per iteration.
// ---------------------------------------------------------------------------
template <int NT, bool K16>
__global__ void gemm_wmma_kernel(const float* __restrict__ A, const float* __restrict__ W,
                                 const float* __restrict__ bias, const float* __restrict__ resid,
                                 float* __restrict__ out,
                                 int M, int N, int K, int lda, int ldw, int ldo, int postop) {
  int wave = blockIdx.x * (blockDim.x >> 5) + (threadIdx.x >> 5);
  int lane = threadIdx.x & 31;
  int groupsN = N / (16 * NT);
  int mt = wave / groupsN;
  int ng = wave - mt * groupsN;
  if (mt * 16 >= M) return;                 // wave-uniform exit (EXEC stays full)
  int r = lane & 15, hh = lane >> 4;

  const float* arow = A + (size_t)(mt * 16 + r) * lda;
  v8f acc[NT];
#pragma unroll
  for (int i = 0; i < NT; ++i) acc[i] = (v8f){};

  if (K16) {
    v16h a = load_frag16(arow, hh);
#pragma unroll
    for (int i = 0; i < NT; ++i) {
      v16h b = load_frag16(W + (size_t)(ng * NT * 16 + i * 16 + r) * ldw, hh);
      acc[i] = __builtin_amdgcn_wmma_f32_16x16x32_f16(false, a, false, b,
                                                      (short)0, acc[i], false, false);
    }
  } else {
    for (int kb = 0; kb < K; kb += 32) {
      v16h a = load_frag32(arow + kb, hh);
#pragma unroll
      for (int i = 0; i < NT; ++i) {
        v16h b = load_frag32(W + (size_t)(ng * NT * 16 + i * 16 + r) * ldw + kb, hh);
        acc[i] = __builtin_amdgcn_wmma_f32_16x16x32_f16(false, a, false, b,
                                                        (short)0, acc[i], false, false);
      }
    }
  }

#pragma unroll
  for (int i = 0; i < NT; ++i) {            // C/D layout: lane->N, vgpr j -> M=j+8*(lane/16)
    int n = ng * NT * 16 + i * 16 + r;
    float bv = bias ? bias[n] : 0.0f;
#pragma unroll
    for (int j = 0; j < 8; ++j) {
      int m = mt * 16 + j + 8 * hh;
      float v = acc[i][j] + bv;
      if (postop == 1) v = (v > 20.0f) ? v : log1pf(__expf(v));
      if (resid) v += resid[(size_t)m * ldo + n];
      out[(size_t)m * ldo + n] = v;
    }
  }
}

static inline void launch_gemm(const float* A, const float* W, const float* bias,
                               const float* resid, float* out,
                               int M, int N, int K, int lda, int ldw, int ldo,
                               int postop, hipStream_t s) {
  if (K == 16) {                            // dt_proj: N=512, K=16 (zero-padded)
    int waves = (M / 16) * (N / 64);
    gemm_wmma_kernel<4, true><<<(waves + 7) / 8, 256, 0, s>>>(
        A, W, bias, resid, out, M, N, K, lda, ldw, ldo, postop);
  } else if (N % 64 == 0) {
    int waves = (M / 16) * (N / 64);
    gemm_wmma_kernel<4, false><<<(waves + 7) / 8, 256, 0, s>>>(
        A, W, bias, resid, out, M, N, K, lda, ldw, ldo, postop);
  } else {                                  // x_proj: N=48
    int waves = (M / 16) * (N / 48);
    gemm_wmma_kernel<3, false><<<(waves + 7) / 8, 256, 0, s>>>(
        A, W, bias, resid, out, M, N, K, lda, ldw, ldo, postop);
  }
}

// ---------------------------------------------------------------------------
// Time-reverse copy: out[b,l,:] = in[b,L-1-l,:]
// ---------------------------------------------------------------------------
__global__ void reverse_time_kernel(const float* __restrict__ in, float* __restrict__ out, int C) {
  size_t idx = (size_t)blockIdx.x * blockDim.x + threadIdx.x;
  if (idx >= (size_t)TOK * C) return;
  int c = (int)(idx % C);
  size_t tok = idx / C;
  int b = (int)(tok / Ldim), l = (int)(tok % Ldim);
  out[idx] = in[((size_t)b * Ldim + (Ldim - 1 - l)) * C + c];
}

// Causal depthwise conv (kernel 4) + bias + SiLU. xi = xz[:, :DIN], ld = 2*DIN.
__global__ void conv_silu_kernel(const float* __restrict__ xz, const float* __restrict__ cw,
                                 const float* __restrict__ cb, float* __restrict__ u) {
  size_t idx = (size_t)blockIdx.x * blockDim.x + threadIdx.x;
  if (idx >= (size_t)TOK * DIN) return;
  int d = (int)(idx & (DIN - 1));
  size_t tok = idx >> 9;
  int l = (int)(tok & (Ldim - 1));
  float acc = cb[d];
#pragma unroll
  for (int j = 0; j < DCONV; ++j) {
    int ls = l - (DCONV - 1) + j;
    if (ls >= 0)
      acc += cw[d * DCONV + j] * xz[(tok - (DCONV - 1) + j) * (2 * DIN) + d];
  }
  u[idx] = acc * (1.0f / (1.0f + __expf(-acc)));
}

// ---------------------------------------------------------------------------
// Chunked selective scan, pass 1: local scan per (b, chunk, d) from zero state.
// Emits per-step local y, chunk-final h[16], and sum(dt) over the chunk.
// B/C rows (dbc cols 16..47) staged in LDS (16 KB) and broadcast-read.
// ---------------------------------------------------------------------------
__global__ void scan_pass1_kernel(const float* __restrict__ dt, const float* __restrict__ u,
                                  const float* __restrict__ dbc, const float* __restrict__ A_log,
                                  float* __restrict__ yscan, float* __restrict__ hloc,
                                  float* __restrict__ sumdt) {
  __shared__ float bc[CL][32];              // [t][0:16]=B, [t][16:32]=C
  int bcid = blockIdx.x;                    // b*NC + c
  int b = bcid / NC, c = bcid % NC;
  int tid = threadIdx.x;                    // d channel (blockDim = 512)
  size_t tok0 = (size_t)b * Ldim + (size_t)c * CL;
  for (int i = tid; i < CL * 32; i += DIN) {
    int t = i >> 5, j = i & 31;
    bc[t][j] = dbc[(tok0 + t) * 48 + 16 + j];
  }
  __syncthreads();

  int d = tid;
  float Arow[DSTATE], h[DSTATE];
#pragma unroll
  for (int s = 0; s < DSTATE; ++s) {
    Arow[s] = -__expf(A_log[d * DSTATE + s]);
    h[s] = 0.0f;
  }
  float sd = 0.0f;
  for (int t = 0; t < CL; ++t) {
    size_t tok = tok0 + t;
    float dtv = dt[tok * DIN + d];
    float du  = dtv * u[tok * DIN + d];
    float y = 0.0f;
#pragma unroll
    for (int s = 0; s < DSTATE; ++s) {
      float e = __expf(dtv * Arow[s]);
      h[s] = e * h[s] + du * bc[t][s];
      y += h[s] * bc[t][16 + s];
    }
    sd += dtv;
    yscan[tok * DIN + d] = y;
  }
  size_t hidx = ((size_t)bcid * DIN + d) * DSTATE;
#pragma unroll
  for (int s = 0; s < DSTATE; ++s) hloc[hidx + s] = h[s];
  sumdt[(size_t)bcid * DIN + d] = sd;
}

// Pass 2: stitch chunk states sequentially (16 steps) per (b,d).
// Chunk decay = exp(A[s] * sum_dt_chunk) since prod exp(dt*A) = exp(A*sum dt).
__global__ void scan_pass2_kernel(const float* __restrict__ A_log, const float* __restrict__ hloc,
                                  const float* __restrict__ sumdt, float* __restrict__ Sin) {
  int idx = blockIdx.x * blockDim.x + threadIdx.x;  // b*DIN + d
  if (idx >= Bdim * DIN) return;
  int b = idx / DIN, d = idx % DIN;
  float Arow[DSTATE], S[DSTATE];
#pragma unroll
  for (int s = 0; s < DSTATE; ++s) {
    Arow[s] = -__expf(A_log[d * DSTATE + s]);
    S[s] = 0.0f;
  }
  for (int c = 0; c < NC; ++c) {
    size_t base = ((size_t)(b * NC + c) * DIN + d) * DSTATE;
#pragma unroll
    for (int s = 0; s < DSTATE; ++s) Sin[base + s] = S[s];
    float sd = sumdt[(size_t)(b * NC + c) * DIN + d];
#pragma unroll
    for (int s = 0; s < DSTATE; ++s) S[s] = __expf(Arow[s] * sd) * S[s] + hloc[base + s];
  }
}

// Pass 3: add carried-state contribution y_t += sum_s C_t[s]*exp(A[s]*cumdt_t)*S_in[s].
// Chunk 0 has zero carry -> launched only for c in [1, NC).
__global__ void scan_pass3_kernel(const float* __restrict__ dt, const float* __restrict__ dbc,
                                  const float* __restrict__ A_log, const float* __restrict__ Sin,
                                  float* __restrict__ yscan) {
  __shared__ float cc[CL][DSTATE];          // C rows, 8 KB
  int bid = blockIdx.x;
  int b = bid / (NC - 1);
  int c = 1 + bid % (NC - 1);
  int tid = threadIdx.x;
  size_t tok0 = (size_t)b * Ldim + (size_t)c * CL;
  for (int i = tid; i < CL * DSTATE; i += DIN) {
    int t = i >> 4, j = i & 15;
    cc[t][j] = dbc[(tok0 + t) * 48 + 32 + j];
  }
  __syncthreads();

  int d = tid;
  float Arow[DSTATE], S[DSTATE];
  size_t sbase = ((size_t)(b * NC + c) * DIN + d) * DSTATE;
#pragma unroll
  for (int s = 0; s < DSTATE; ++s) {
    Arow[s] = -__expf(A_log[d * DSTATE + s]);
    S[s] = Sin[sbase + s];
  }
  float cum = 0.0f;
  for (int t = 0; t < CL; ++t) {
    size_t tok = tok0 + t;
    cum += dt[tok * DIN + d];
    float y = 0.0f;
#pragma unroll
    for (int s = 0; s < DSTATE; ++s) y += cc[t][s] * __expf(Arow[s] * cum) * S[s];
    yscan[tok * DIN + d] += y;
  }
}

// y = (y_scan + u*D) * silu(z);  z = xz[:, DIN + d]
__global__ void gate_kernel(const float* __restrict__ u, const float* __restrict__ xz,
                            const float* __restrict__ Dw, float* __restrict__ yscan) {
  size_t idx = (size_t)blockIdx.x * blockDim.x + threadIdx.x;
  if (idx >= (size_t)TOK * DIN) return;
  int d = (int)(idx & (DIN - 1));
  size_t tok = idx >> 9;
  float z = xz[tok * (2 * DIN) + DIN + d];
  float g = z * (1.0f / (1.0f + __expf(-z)));
  yscan[idx] = (yscan[idx] + u[idx] * Dw[d]) * g;
}

// adjT[j,i] = softmax_j(relu(E @ E^T))[i,j]  (transposed so GEMM uses W[n,k] form)
__global__ void adjT_kernel(const float* __restrict__ node_emb, float* __restrict__ adjT) {
  __shared__ float E[Hdim][10];
  int i = threadIdx.x;
  for (int k = 0; k < 10; ++k) E[i][k] = node_emb[i * 10 + k];
  __syncthreads();
  float e[10];
#pragma unroll
  for (int k = 0; k < 10; ++k) e[k] = E[i][k];
  float mx = -1e30f;
  for (int j = 0; j < Hdim; ++j) {
    float g = 0.0f;
#pragma unroll
    for (int k = 0; k < 10; ++k) g += e[k] * E[j][k];
    mx = fmaxf(mx, fmaxf(g, 0.0f));
  }
  float sum = 0.0f;
  for (int j = 0; j < Hdim; ++j) {
    float g = 0.0f;
#pragma unroll
    for (int k = 0; k < 10; ++k) g += e[k] * E[j][k];
    sum += __expf(fmaxf(g, 0.0f) - mx);
  }
  float inv = 1.0f / sum;
  for (int j = 0; j < Hdim; ++j) {
    float g = 0.0f;
#pragma unroll
    for (int k = 0; k < 10; ++k) g += e[k] * E[j][k];
    adjT[j * Hdim + i] = __expf(fmaxf(g, 0.0f) - mx) * inv;
  }
}

// x_time[b,l,h] = out_f[b,l,h] + out_b[b, L-1-l, h]
__global__ void combine_kernel(const float* __restrict__ f, const float* __restrict__ bw,
                               float* __restrict__ out) {
  size_t idx = (size_t)blockIdx.x * blockDim.x + threadIdx.x;
  if (idx >= (size_t)TOK * Hdim) return;
  int h = (int)(idx & (Hdim - 1));
  size_t tok = idx >> 8;
  int b = (int)(tok / Ldim), l = (int)(tok % Ldim);
  out[idx] = f[idx] + bw[((size_t)b * Ldim + (Ldim - 1 - l)) * Hdim + h];
}

// LayerNorm over H=256 per row, block-per-row with LDS tree reduction.
__global__ void layernorm_kernel(const float* __restrict__ xg, const float* __restrict__ w,
                                 const float* __restrict__ bvec, float* __restrict__ out) {
  __shared__ float red[Hdim];
  int row = blockIdx.x, t = threadIdx.x;
  float v = xg[(size_t)row * Hdim + t];
  red[t] = v;
  __syncthreads();
  for (int s = Hdim / 2; s > 0; s >>= 1) {
    if (t < s) red[t] += red[t + s];
    __syncthreads();
  }
  float mu = red[0] * (1.0f / Hdim);
  __syncthreads();
  float dv = v - mu;
  red[t] = dv * dv;
  __syncthreads();
  for (int s = Hdim / 2; s > 0; s >>= 1) {
    if (t < s) red[t] += red[t + s];
    __syncthreads();
  }
  float var = red[0] * (1.0f / Hdim);
  out[(size_t)row * Hdim + t] = dv * rsqrtf(var + 1e-5f) * w[t] + bvec[t];
}

// ---------------------------------------------------------------------------
// Host orchestration. All launches on `stream`; scratch carved from d_ws.
// ws usage: ~66M floats (~264 MB).
// ---------------------------------------------------------------------------
extern "C" void kernel_launch(void* const* d_in, const int* in_sizes, int n_in,
                              void* d_out, int out_size, void* d_ws, size_t ws_size,
                              hipStream_t stream) {
  (void)in_sizes; (void)n_in; (void)out_size; (void)ws_size;
  const float* x        = (const float*)d_in[0];
  const float* emb_w    = (const float*)d_in[1];
  const float* emb_b    = (const float*)d_in[2];
  const float* node_emb = (const float*)d_in[21];
  const float* ln_w     = (const float*)d_in[22];
  const float* ln_b     = (const float*)d_in[23];

  float* ws = (float*)d_ws;
  size_t off = 0;
  auto carve = [&](size_t n) { float* p = ws + off; off += n; return p; };
  float* xe     = carve((size_t)TOK * Hdim);
  float* xe_rev = carve((size_t)TOK * Hdim);
  float* out_f  = carve((size_t)TOK * Hdim);
  float* out_b  = carve((size_t)TOK * Hdim);
  float* x_time = carve((size_t)TOK * Hdim);
  float* xz     = carve((size_t)TOK * 2 * DIN);
  float* ubuf   = carve((size_t)TOK * DIN);
  float* dbc    = carve((size_t)TOK * 48);
  float* dtb    = carve((size_t)TOK * DIN);
  float* yscan  = carve((size_t)TOK * DIN);
  float* hloc   = carve((size_t)Bdim * NC * DIN * DSTATE);
  float* sumdt  = carve((size_t)Bdim * NC * DIN);
  float* Sin    = carve((size_t)Bdim * NC * DIN * DSTATE);
  float* adjT   = carve((size_t)Hdim * Hdim);

  // 1) Embedding: xe = x @ emb_w^T + emb_b
  launch_gemm(x, emb_w, emb_b, nullptr, xe, TOK, Hdim, Fdim, Fdim, Fdim, Hdim, 0, stream);

  // 2) Reversed copy for the backward direction
  {
    size_t n = (size_t)TOK * Hdim;
    reverse_time_kernel<<<(unsigned)((n + 255) / 256), 256, 0, stream>>>(xe, xe_rev, Hdim);
  }

  // 3) Both Mamba directions (buffers reused; launches serialized on stream)
  for (int dir = 0; dir < 2; ++dir) {
    int base = 3 + dir * 9;
    const float* in_proj_w  = (const float*)d_in[base + 0];
    const float* conv_w     = (const float*)d_in[base + 1];
    const float* conv_b     = (const float*)d_in[base + 2];
    const float* x_proj_w   = (const float*)d_in[base + 3];
    const float* dt_proj_w  = (const float*)d_in[base + 4];
    const float* dt_proj_b  = (const float*)d_in[base + 5];
    const float* A_log      = (const float*)d_in[base + 6];
    const float* Dw         = (const float*)d_in[base + 7];
    const float* out_proj_w = (const float*)d_in[base + 8];
    const float* src = dir ? xe_rev : xe;
    float* dst = dir ? out_b : out_f;

    launch_gemm(src, in_proj_w, nullptr, nullptr, xz,
                TOK, 2 * DIN, Hdim, Hdim, Hdim, 2 * DIN, 0, stream);
    {
      size_t n = (size_t)TOK * DIN;
      conv_silu_kernel<<<(unsigned)((n + 255) / 256), 256, 0, stream>>>(xz, conv_w, conv_b, ubuf);
    }
    launch_gemm(ubuf, x_proj_w, nullptr, nullptr, dbc, TOK, 48, DIN, DIN, DIN, 48, 0, stream);
    launch_gemm(dbc, dt_proj_w, dt_proj_b, nullptr, dtb,
                TOK, DIN, DTRANK, 48, DTRANK, DIN, /*softplus*/ 1, stream);

    scan_pass1_kernel<<<Bdim * NC, DIN, 0, stream>>>(dtb, ubuf, dbc, A_log, yscan, hloc, sumdt);
    scan_pass2_kernel<<<(Bdim * DIN + 255) / 256, 256, 0, stream>>>(A_log, hloc, sumdt, Sin);
    scan_pass3_kernel<<<Bdim * (NC - 1), DIN, 0, stream>>>(dtb, dbc, A_log, Sin, yscan);

    {
      size_t n = (size_t)TOK * DIN;
      gate_kernel<<<(unsigned)((n + 255) / 256), 256, 0, stream>>>(ubuf, xz, Dw, yscan);
    }
    launch_gemm(yscan, out_proj_w, nullptr, nullptr, dst, TOK, Hdim, DIN, DIN, DIN, Hdim, 0, stream);
  }

  // 4) Combine directions, graph mixing, LayerNorm
  {
    size_t n = (size_t)TOK * Hdim;
    combine_kernel<<<(unsigned)((n + 255) / 256), 256, 0, stream>>>(out_f, out_b, x_time);
  }
  adjT_kernel<<<1, Hdim, 0, stream>>>(node_emb, adjT);
  // xg = x_time @ adj + x_time  (out_f reused as xg buffer)
  launch_gemm(x_time, adjT, nullptr, x_time, out_f, TOK, Hdim, Hdim, Hdim, Hdim, Hdim, 0, stream);
  layernorm_kernel<<<TOK, Hdim, 0, stream>>>(out_f, ln_w, ln_b, (float*)d_out);
}
